// GNNModel_30142080483987
// MI455X (gfx1250) — compile-verified, matrix operand order checked by reading, not run on hardware
//
#include <hip/hip_runtime.h>
#include <hip/hip_bf16.h>

typedef __attribute__((ext_vector_type(16))) _Float16 v16h;
typedef __attribute__((ext_vector_type(8)))  float    v8f;

static __host__ __device__ inline int cdiv(int a, int b) { return (a + b - 1) / b; }

// ---------------- small utility kernels ----------------

__global__ void k_fill(float* p, float v, int n) {
    int i = blockIdx.x * blockDim.x + threadIdx.x;
    if (i < n) p[i] = v;
}

__global__ void k_deg_edges(float* deg, const long long* __restrict__ col, int E) {
    int e = blockIdx.x * blockDim.x + threadIdx.x;
    if (e < E) atomicAdd(&deg[(int)col[e]], 1.0f);
}

__global__ void k_rsqrt_inplace(float* p, int n) {
    int i = blockIdx.x * blockDim.x + threadIdx.x;
    if (i < n) p[i] = rsqrtf(p[i]);   // deg >= 1 always (self-loop)
}

// ---------------- WMMA GEMM: Y[M,NACT] = X[M,K] * W[K,NACT] ----------------
// f32 inputs converted to f16, f32 accumulation via v_wmma_f32_16x16x32_f16.
// Each wave computes a 16-row x (NTILES*16)-col strip. Block = 8 waves = 128 rows.
// Tail rows are handled by CLAMPING the A row index (row i of Y depends only on
// row i of X, so clamped-garbage rows are never stored) -> unconditional
// vectorized b128 loads, no exec-mask churn in the hot loop.
template<int K, int NTILES, int NACT>
__global__ __launch_bounds__(256)
void k_gemm_wmma(const float* __restrict__ X, const float* __restrict__ W,
                 float* __restrict__ Y, int M) {
    constexpr int KC = K / 32;
    __shared__ v16h Bs[KC][NTILES][32];

    // Cooperative fill of B tiles in the WMMA B-operand lane layout.
    for (int idx = threadIdx.x; idx < KC * NTILES * 32; idx += 256) {
        int kc   = idx / (NTILES * 32);
        int rem  = idx % (NTILES * 32);
        int nt   = rem / 32;
        int lane = rem % 32;
        int L = lane & 15, hi = lane >> 4;
        v16h tmp;
#pragma unroll
        for (int e = 0; e < 16; ++e) {
            int klocal = (e < 8) ? (hi * 8 + e) : (16 + hi * 8 + (e - 8));
            int k = kc * 32 + klocal;
            int n = nt * 16 + L;
            float wv = (n < NACT) ? W[k * NACT + n] : 0.0f;
            tmp[e] = (_Float16)wv;
        }
        Bs[kc][nt][lane] = tmp;
    }
    __syncthreads();

    int wave = threadIdx.x >> 5;
    int lane = threadIdx.x & 31;
    int L = lane & 15, hi = lane >> 4;
    int row0 = blockIdx.x * 128 + wave * 16;
    int ar = row0 + L;
    if (ar > M - 1) ar = M - 1;         // clamp: safe, tail outputs not stored
    const float* xrow = X + (long)ar * K;

    v8f zero = {};
    v8f acc[NTILES];
#pragma unroll
    for (int nt = 0; nt < NTILES; ++nt) acc[nt] = zero;

#pragma unroll
    for (int kc = 0; kc < KC; ++kc) {
        const float* xr = xrow + kc * 32 + hi * 8;   // 32-byte aligned
        float4 f0 = *(const float4*)(xr);
        float4 f1 = *(const float4*)(xr + 4);
        float4 f2 = *(const float4*)(xr + 16);
        float4 f3 = *(const float4*)(xr + 20);
        v16h a;
        a[0]  = (_Float16)f0.x; a[1]  = (_Float16)f0.y;
        a[2]  = (_Float16)f0.z; a[3]  = (_Float16)f0.w;
        a[4]  = (_Float16)f1.x; a[5]  = (_Float16)f1.y;
        a[6]  = (_Float16)f1.z; a[7]  = (_Float16)f1.w;
        a[8]  = (_Float16)f2.x; a[9]  = (_Float16)f2.y;
        a[10] = (_Float16)f2.z; a[11] = (_Float16)f2.w;
        a[12] = (_Float16)f3.x; a[13] = (_Float16)f3.y;
        a[14] = (_Float16)f3.z; a[15] = (_Float16)f3.w;
#pragma unroll
        for (int nt = 0; nt < NTILES; ++nt) {
            acc[nt] = __builtin_amdgcn_wmma_f32_16x16x32_f16(
                false, a, false, Bs[kc][nt][lane], (short)0, acc[nt], false, false);
        }
    }

    // C/D layout: lanes 0-15: VGPR r -> (M=r, N=lane); lanes 16-31: (M=8+r, N=lane-16)
#pragma unroll
    for (int nt = 0; nt < NTILES; ++nt) {
#pragma unroll
        for (int r = 0; r < 8; ++r) {
            int row = row0 + r + 8 * hi;
            int col = nt * 16 + L;
            if (row < M && col < NACT) Y[(long)row * NACT + col] = acc[nt][r];
        }
    }
}

// ---------------- edge gather/scale/scatter-add ----------------
template<int F>
__global__ void k_scatter_edges(const float* __restrict__ hw,
                                const long long* __restrict__ src,
                                const long long* __restrict__ dst,
                                const float* __restrict__ dinv,
                                float* __restrict__ agg, int E) {
    int gid = blockIdx.x * blockDim.x + threadIdx.x;
    int e = gid / F;
    if (e >= E) return;
    int f = gid - e * F;
    int s = (int)src[e];
    int d = (int)dst[e];
    float nrm = dinv[s] * dinv[d];
    atomicAdd(&agg[(long)d * F + f], nrm * hw[(long)s * F + f]);
}

// agg = relu(agg + dinv^2 * hw + b)   (self-loop folded in)
template<int F>
__global__ void k_finalize_relu(float* __restrict__ agg, const float* __restrict__ hw,
                                const float* __restrict__ dinv, const float* __restrict__ b,
                                int M) {
    int gid = blockIdx.x * blockDim.x + threadIdx.x;
    if (gid >= M * F) return;
    int row = gid / F;
    int col = gid - row * F;
    float d2 = dinv[row] * dinv[row];
    float v = agg[gid] + d2 * hw[gid] + b[col];
    agg[gid] = fmaxf(v, 0.0f);
}

// out = softmax(agg + dinv^2*hw + b) over F columns; one wave32 per row, 2 cols/lane.
template<int F>
__global__ __launch_bounds__(256)
void k_softmax_out(const float* __restrict__ agg, const float* __restrict__ hw,
                   const float* __restrict__ dinv, const float* __restrict__ b,
                   float* __restrict__ out, int M) {
    int wave = threadIdx.x >> 5;
    int lane = threadIdx.x & 31;
    int row = blockIdx.x * 8 + wave;
    if (row >= M) return;                 // uniform per wave
    float d2 = dinv[row] * dinv[row];
    int c0 = lane, c1 = lane + 32;
    float v0 = -INFINITY, v1 = -INFINITY;
    if (c0 < F) v0 = agg[(long)row * F + c0] + d2 * hw[(long)row * F + c0] + b[c0];
    if (c1 < F) v1 = agg[(long)row * F + c1] + d2 * hw[(long)row * F + c1] + b[c1];
    float m = fmaxf(v0, v1);
#pragma unroll
    for (int o = 16; o > 0; o >>= 1) m = fmaxf(m, __shfl_xor(m, o, 32));
    float e0 = (c0 < F) ? __expf(v0 - m) : 0.0f;
    float e1 = (c1 < F) ? __expf(v1 - m) : 0.0f;
    float s = e0 + e1;
#pragma unroll
    for (int o = 16; o > 0; o >>= 1) s += __shfl_xor(s, o, 32);
    float inv = 1.0f / s;
    if (c0 < F) out[(long)row * F + c0] = e0 * inv;
    if (c1 < F) out[(long)row * F + c1] = e1 * inv;
}

// ---------------- launch ----------------

extern "C" void kernel_launch(void* const* d_in, const int* in_sizes, int n_in,
                              void* d_out, int out_size, void* d_ws, size_t ws_size,
                              hipStream_t stream) {
    const float*      x  = (const float*)d_in[0];       // [N,128]
    const long long*  ei = (const long long*)d_in[1];   // [2,E] int64
    const float*      W1 = (const float*)d_in[2];       // [128,64]
    const float*      b1 = (const float*)d_in[3];       // [64]
    const float*      W2 = (const float*)d_in[4];       // [64,40]
    const float*      b2 = (const float*)d_in[5];       // [40]
    float*            out = (float*)d_out;              // [N,40]

    const int N = in_sizes[0] / 128;
    const int E = in_sizes[1] / 2;
    const long long* src = ei;        // row (source)
    const long long* dst = ei + E;    // col (target, aggregation axis)

    // workspace layout (floats)
    float* wsF  = (float*)d_ws;
    float* dinv = wsF;                         // [N]       (deg -> rsqrt in place)
    float* hw1  = dinv + 50048;                // [N,64]
    float* agg1 = hw1 + (long)N * 64;          // [N,64]    (becomes h after relu)
    float* hw2  = agg1 + (long)N * 64;         // [N,40]
    float* agg2 = hw2 + (long)N * 40;          // [N,40]

    const int TB = 256;

    // 1) degrees & symmetric norm factors
    k_fill<<<cdiv(N, TB), TB, 0, stream>>>(dinv, 1.0f, N);             // self-loop
    k_deg_edges<<<cdiv(E, TB), TB, 0, stream>>>(dinv, dst, E);
    k_rsqrt_inplace<<<cdiv(N, TB), TB, 0, stream>>>(dinv, N);

    // zero accumulators
    k_fill<<<cdiv(N * 64, TB), TB, 0, stream>>>(agg1, 0.0f, N * 64);
    k_fill<<<cdiv(N * 40, TB), TB, 0, stream>>>(agg2, 0.0f, N * 40);

    // 2) layer 1: hw1 = x @ W1 (WMMA), scatter, relu(agg + d2*hw + b)
    k_gemm_wmma<128, 4, 64><<<cdiv(N, 128), 256, 0, stream>>>(x, W1, hw1, N);
    k_scatter_edges<64><<<cdiv(E * 64, TB), TB, 0, stream>>>(hw1, src, dst, dinv, agg1, E);
    k_finalize_relu<64><<<cdiv(N * 64, TB), TB, 0, stream>>>(agg1, hw1, dinv, b1, N);

    // 3) layer 2: hw2 = h @ W2 (WMMA, N=40 padded to 48), scatter, softmax
    k_gemm_wmma<64, 3, 40><<<cdiv(N, 128), 256, 0, stream>>>(agg1, W2, hw2, N);
    k_scatter_edges<40><<<cdiv(E * 40, TB), TB, 0, stream>>>(hw2, src, dst, dinv, agg2, E);
    k_softmax_out<40><<<cdiv(N, 8), 256, 0, stream>>>(agg2, hw2, dinv, b2, out, N);
}
